// Downsample_PASA_group_softmax_22144851378639
// MI455X (gfx1250) — matrix operand order repaired and checked by reading, not compiled, not run on hardware
//
#include <hip/hip_runtime.h>
#include <hip/hip_bf16.h>

// ---------------------------------------------------------------------------
// PASA group-softmax dynamic pooling:
//   sigma = softmax_ch( BN( conv3x3_reflect(x, w) ) )   (18 = G*K*K channels)
//   out[n,c,y,x] = sum_{r,s} x_reflpad[n,c,y+r,x+s] * sigma[n, g(c), r*3+s, y, x]
// Kernel 1: implicit-GEMM conv on v_wmma_f32_16x16x32_bf16. Both operands are
//   staged in LDS in WMMA *fragment element order*, so each lane fetches its
//   16 bf16 operand values with 2x ds_load_b128 (no scalar u16 gather/pack).
// Kernel 2: bandwidth-bound 9-tap dynamic pooling (the HBM roofline term).
// ---------------------------------------------------------------------------

typedef __bf16 v8bf  __attribute__((ext_vector_type(8)));
typedef __bf16 v16bf __attribute__((ext_vector_type(16)));
typedef float  v8f   __attribute__((ext_vector_type(8)));

#define NIMG 8
#define C    256
#define H    128
#define W    128
#define OC   18          // G*K*K
#define OCP  32          // padded to 2 WMMA N-tiles
#define KDIM 3
#define EPSV 1e-5f

#define CI_CHUNK 32                  // input channels staged per LDS pass
#define NCHUNK   (C / CI_CHUNK)      // 8
#define KLOC     (CI_CHUNK * 9)      // 288 k-values per chunk
#define KSTEPS   (KLOC / 32)         // 9 WMMA k-steps per chunk

#define PXB        64                // pixels (GEMM M rows) per block
#define IMC_STRIDE 296               // im2col row stride (bf16); mult of 8 -> 16B-aligned runs

__device__ __forceinline__ int refl(int i, int n) {
  if (i < 0) i = -i;
  if (i >= n) i = 2 * n - 2 - i;
  return i;
}

// ---------------------------------------------------------------------------
// Kernel 1: conv(256->18, 3x3 reflect) + BN(eval) + softmax over 18 channels.
// Grid: (H * W/PXB, NIMG); block: 128 threads = 4 waves; wave w owns pixels
// px = 16w .. 16w+15 of the block's 64-pixel strip.
// ---------------------------------------------------------------------------
__global__ __launch_bounds__(128)
void pasa_conv_bn_softmax(const float* __restrict__ x,
                          const float* __restrict__ cw,
                          const float* __restrict__ gamma,
                          const float* __restrict__ beta,
                          const float* __restrict__ rmean,
                          const float* __restrict__ rvar,
                          float* __restrict__ sigma) {
  // im2col slab, row-per-pixel contiguous in k -> A fragments are two
  // 16B-aligned 8-value runs per lane.
  __shared__ __attribute__((aligned(16))) __bf16 imc[PXB * IMC_STRIDE];   // 37,888 B
  // weight tile pre-permuted to fragment order: [(ks*2+nt)*32 + lane][16]
  // -> each lane's B fragment is one 32B-aligned 32-byte run.
  __shared__ __attribute__((aligned(32))) __bf16 wtf[KSTEPS * 2 * 32 * 16]; // 18,432 B
  __shared__ float st[PXB * OCP];                                          //  8,192 B

  const int tid  = threadIdx.x;
  const int y    = blockIdx.x >> 1;
  const int x0   = (blockIdx.x & 1) * PXB;
  const int img  = blockIdx.y;
  const int wave = tid >> 5;
  const int lane = tid & 31;
  const int mrow = lane & 15;   // M row (pixel) / N column within tile
  const int hi   = lane >> 4;   // lane half selects K sub-range per ISA layout

  v8f acc0 = {};  // N = 0..15
  v8f acc1 = {};  // N = 16..31 (only 16,17 valid)

  const int gy0 = refl(y - 1, H), gy1 = y, gy2 = refl(y + 1, H);

  for (int cc = 0; cc < NCHUNK; ++cc) {
    __syncthreads();  // previous iteration's fragment reads are done

    // --- Stage im2col: imc[px][k], k = cil*9 + r*3 + s.  Pixel-fastest
    //     iteration keeps the global f32 reads coalesced.
    for (int idx = tid; idx < KLOC * PXB; idx += 128) {
      int k   = idx >> 6;          // / PXB
      int px  = idx & (PXB - 1);
      int cil = k / 9;
      int t   = k - cil * 9;
      int r   = t / 3;
      int s   = t - r * 3;
      int gy  = (r == 0) ? gy0 : ((r == 1) ? gy1 : gy2);
      int gx  = refl(x0 + px - 1 + s, W);
      imc[px * IMC_STRIDE + k] =
          (__bf16)x[(((size_t)img * C + cc * CI_CHUNK + cil) * H + gy) * W + gx];
    }
    // --- Stage weights in fragment order: slot (ks, nt, L) holds the 16
    //     bf16 values lane L feeds to WMMA: K = ks*32 + (e>>3)*16 + (L>>4)*8 + (e&7),
    //     oc = nt*16 + (L&15) (zero-padded past 18).
    for (int idx = tid; idx < KSTEPS * 2 * 32 * 16; idx += 128) {
      int e    = idx & 15;
      int fi   = idx >> 4;
      int L    = fi & 31;
      int nt   = (fi >> 5) & 1;
      int ks   = fi >> 6;
      int k    = ks * 32 + (e >> 3) * 16 + (L >> 4) * 8 + (e & 7);
      int oc   = nt * 16 + (L & 15);
      int cil  = k / 9;
      int t    = k - cil * 9;
      float wv = 0.f;
      if (oc < OC)
        wv = cw[(((size_t)oc * C + cc * CI_CHUNK + cil) * KDIM + t / 3) * KDIM + (t % 3)];
      wtf[idx] = (__bf16)wv;
    }
    __syncthreads();

    const __bf16* arow = &imc[(wave * 16 + mrow) * IMC_STRIDE];
#pragma unroll
    for (int ks = 0; ks < KSTEPS; ++ks) {
      // A fragment: two contiguous 8-value runs (ds_load_b128 each).
      v8bf alo = *(const v8bf*)(arow + ks * 32 + hi * 8);
      v8bf ahi = *(const v8bf*)(arow + ks * 32 + 16 + hi * 8);
      v16bf a  = __builtin_shufflevector(alo, ahi, 0, 1, 2, 3, 4, 5, 6, 7,
                                         8, 9, 10, 11, 12, 13, 14, 15);
      // B fragments: one contiguous 32-byte run per lane (2x ds_load_b128).
      v16bf b0 = *(const v16bf*)&wtf[(ks * 64 + lane) * 16];
      v16bf b1 = *(const v16bf*)&wtf[(ks * 64 + 32 + lane) * 16];
      acc0 = __builtin_amdgcn_wmma_f32_16x16x32_bf16(false, a, false, b0,
                                                     (short)0, acc0, false, false);
      acc1 = __builtin_amdgcn_wmma_f32_16x16x32_bf16(false, a, false, b1,
                                                     (short)0, acc1, false, false);
    }
  }

  __syncthreads();
  // D layout: VGPR i holds M = i + 8*hi, column N = lane & 15.
#pragma unroll
  for (int i = 0; i < 8; ++i) {
    int prow = wave * 16 + i + hi * 8;
    st[prow * OCP + mrow]      = acc0[i];
    st[prow * OCP + 16 + mrow] = acc1[i];
  }
  __syncthreads();

  // BN (eval) + softmax over the 18 valid channels; coalesced plane writes.
  if (tid < PXB) {
    int   px = tid;
    float v[OC];
    float mx = -3.4e38f;
#pragma unroll
    for (int oc = 0; oc < OC; ++oc) {
      float sc = gamma[oc] * rsqrtf(rvar[oc] + EPSV);
      float sh = beta[oc] - rmean[oc] * sc;
      float t  = st[px * OCP + oc] * sc + sh;
      v[oc] = t;
      mx    = fmaxf(mx, t);
    }
    float sum = 0.f;
#pragma unroll
    for (int oc = 0; oc < OC; ++oc) {
      float e = __expf(v[oc] - mx);
      v[oc] = e;
      sum  += e;
    }
    float inv = 1.f / sum;
#pragma unroll
    for (int oc = 0; oc < OC; ++oc)
      sigma[(((size_t)img * OC + oc) * H + y) * W + x0 + px] = v[oc] * inv;
  }
}

// ---------------------------------------------------------------------------
// Kernel 2: per-pixel dynamic 3x3 pooling (bandwidth-bound).
// Grid: (H, NIMG, C/CPB); block: 128 threads = one image row (coalesced).
// ---------------------------------------------------------------------------
#define CPB 8  // channels per block (all within one group: 128 ch per group)

__global__ __launch_bounds__(128)
void pasa_pool(const float* __restrict__ x,
               const float* __restrict__ sigma,
               float* __restrict__ out) {
  const int px  = threadIdx.x;
  const int y   = blockIdx.x;
  const int img = blockIdx.y;
  const int c0  = blockIdx.z * CPB;
  const int g   = c0 / (C / 2);

  int yy[3], xx[3];
#pragma unroll
  for (int r = 0; r < 3; ++r) yy[r] = refl(y - 1 + r, H);
#pragma unroll
  for (int s = 0; s < 3; ++s) xx[s] = refl(px - 1 + s, W);

  float s9[9];
#pragma unroll
  for (int kk = 0; kk < 9; ++kk)
    s9[kk] = sigma[(((size_t)img * OC + g * 9 + kk) * H + y) * W + px];

#pragma unroll
  for (int c = 0; c < CPB; ++c) {
    const float* xc = x + ((size_t)img * C + c0 + c) * H * W;
    if (c + 1 < CPB) {  // software prefetch of the next channel's rows
      const float* xn = xc + (size_t)H * W;
      __builtin_prefetch(xn + (size_t)yy[0] * W + px, 0, 3);
      __builtin_prefetch(xn + (size_t)yy[1] * W + px, 0, 3);
      __builtin_prefetch(xn + (size_t)yy[2] * W + px, 0, 3);
    }
    float acc = 0.f;
#pragma unroll
    for (int r = 0; r < 3; ++r) {
      const float* row = xc + (size_t)yy[r] * W;
#pragma unroll
      for (int s = 0; s < 3; ++s)
        acc = fmaf(row[xx[s]], s9[r * 3 + s], acc);
    }
    out[((size_t)img * C + c0 + c) * H * W + (size_t)y * W + px] = acc;
  }
}

// ---------------------------------------------------------------------------
extern "C" void kernel_launch(void* const* d_in, const int* in_sizes, int n_in,
                              void* d_out, int out_size, void* d_ws, size_t ws_size,
                              hipStream_t stream) {
  const float* x     = (const float*)d_in[0];
  const float* cw    = (const float*)d_in[1];
  const float* gamma = (const float*)d_in[2];
  const float* beta  = (const float*)d_in[3];
  const float* rmean = (const float*)d_in[4];
  const float* rvar  = (const float*)d_in[5];
  float* out   = (float*)d_out;
  float* sigma = (float*)d_ws;  // needs NIMG*OC*H*W*4 = 9.4 MB of scratch

  pasa_conv_bn_softmax<<<dim3(H * (W / PXB), NIMG), 128, 0, stream>>>(
      x, cw, gamma, beta, rmean, rvar, sigma);
  pasa_pool<<<dim3(H, NIMG, C / CPB), 128, 0, stream>>>(x, sigma, out);
}
